// AlignConLoss_72756745994685
// MI455X (gfx1250) — matrix operand order repaired and checked by reading, not compiled, not run on hardware
//
#include <hip/hip_runtime.h>

// ---------------------------------------------------------------------------
// AlignConLoss for MI455X (gfx1250, wave32, WMMA).
//
//   sim[i,j] = <c_i / |c_i| , a_j / |a_j|>     (c = embedding2, a = embedding1)
//   loss     = sum_j ( logsumexp_i sim[i,j] - sim[j,j] )
//
// Fused GEMM + online softmax; the 8192x8192 sim matrix is never
// materialized. FP16 WMMA with F32 accumulate. Normalized f16 operands
// (8 MB total) live in workspace and are fully L2-resident (192 MB L2),
// so phase-2 global traffic is all L2 hits. Each block owns 32 columns
// (two register-resident B tiles) so every A fragment load feeds TWO
// v_wmma_f32_16x16x32_f16 ops -> halves L2 traffic vs. 16-column tiling.
// ---------------------------------------------------------------------------

typedef __attribute__((ext_vector_type(16))) _Float16 v16h;
typedef __attribute__((ext_vector_type(8)))  _Float16 h8;
typedef __attribute__((ext_vector_type(8)))  float    v8f;
typedef __attribute__((ext_vector_type(4)))  float    f4;

#define B_ROWS 8192
#define DDIM   256

// ---------------------------------------------------------------------------
// Phase 1: row L2-normalize both inputs -> f16, plus diag[j] = cos(a_j, c_j).
// One wave (32 lanes) per row; 8 floats per lane.
// ---------------------------------------------------------------------------
__global__ void __launch_bounds__(256)
normalize_kernel(const float* __restrict__ e1,   // anchors  a
                 const float* __restrict__ e2,   // contrast c
                 _Float16* __restrict__ aH,
                 _Float16* __restrict__ cH,
                 float* __restrict__ diag)
{
    const int wid  = threadIdx.x >> 5;
    const int lane = threadIdx.x & 31;
    const int row  = blockIdx.x * 8 + wid;

    const float* pa = e1 + (size_t)row * DDIM + lane * 8;
    const float* pc = e2 + (size_t)row * DDIM + lane * 8;
    f4 a0 = *(const f4*)(pa);
    f4 a1 = *(const f4*)(pa + 4);
    f4 c0 = *(const f4*)(pc);
    f4 c1 = *(const f4*)(pc + 4);

    float ssa = 0.f, ssc = 0.f, dot = 0.f;
#pragma unroll
    for (int k = 0; k < 4; ++k) {
        ssa += a0[k] * a0[k] + a1[k] * a1[k];
        ssc += c0[k] * c0[k] + c1[k] * c1[k];
        dot += a0[k] * c0[k] + a1[k] * c1[k];
    }
    // wave32 butterfly reduction
#pragma unroll
    for (int off = 16; off >= 1; off >>= 1) {
        ssa += __shfl_xor(ssa, off, 32);
        ssc += __shfl_xor(ssc, off, 32);
        dot += __shfl_xor(dot, off, 32);
    }
    const float inva = 1.0f / fmaxf(sqrtf(ssa), 1e-8f);
    const float invc = 1.0f / fmaxf(sqrtf(ssc), 1e-8f);

    h8 ah, ch;
#pragma unroll
    for (int k = 0; k < 4; ++k) {
        ah[k]     = (_Float16)(a0[k] * inva);
        ah[k + 4] = (_Float16)(a1[k] * inva);
        ch[k]     = (_Float16)(c0[k] * invc);
        ch[k + 4] = (_Float16)(c1[k] * invc);
    }
    *(h8*)(aH + (size_t)row * DDIM + lane * 8) = ah;
    *(h8*)(cH + (size_t)row * DDIM + lane * 8) = ch;

    if (lane == 0) diag[row] = dot * inva * invc;   // sim[j,j]
}

// ---------------------------------------------------------------------------
// Phase 2: per 32-column block, stream all 8192 rows through WMMA with an
// online softmax (max / scaled-sum) kept lane-local: C/D layout -> lane L's
// 8 accum VGPRs are column N=(L&15), rows M = v + 8*(L>=16).  8 waves
// stride over row tiles; both B operands (2 x 16 anchors x 256) are
// register-resident (16 x v16h); each A fragment feeds two WMMAs.
// ---------------------------------------------------------------------------
__global__ void __launch_bounds__(256)
simloss_kernel(const _Float16* __restrict__ aH,
               const _Float16* __restrict__ cH,
               const float* __restrict__ diag,
               float* __restrict__ partial)
{
    const int tid  = threadIdx.x;
    const int wid  = tid >> 5;
    const int lane = tid & 31;
    const int col  = lane & 15;      // N index owned by this lane
    const int hi   = lane >> 4;      // lane half (K-group select)
    const int j0   = blockIdx.x * 32;

    // B fragments for two column tiles (a rows j0..j0+31, all of K=256).
    // Contiguous K order used here (matches A): lane L half h -> B[K = 16*(L>=16)+h][N = L&15]
    // == 32 contiguous bytes in row-major aH.
    v16h bfrag0[8], bfrag1[8];
    {
        const _Float16* b0 = aH + (size_t)(j0 + col) * DDIM + hi * 16;
        const _Float16* b1 = b0 + (size_t)16 * DDIM;
#pragma unroll
        for (int kc = 0; kc < 8; ++kc) {
            bfrag0[kc] = *(const v16h*)(b0 + kc * 32);
            bfrag1[kc] = *(const v16h*)(b1 + kc * 32);
        }
    }

    float m0 = -1e30f, s0 = 0.f;   // online softmax state, column j0+col
    float m1 = -1e30f, s1 = 0.f;   // online softmax state, column j0+16+col

    for (int it = wid; it < B_ROWS / 16; it += 8) {
        const int i0 = it * 16;
        const _Float16* pA = cH + (size_t)(i0 + col) * DDIM;

        v8f acc0 = {};
        v8f acc1 = {};
#pragma unroll
        for (int kc = 0; kc < 8; ++kc) {
            // A layout (matches B): lane L halves 0-15 -> K = 16*(L>=16)+h:
            // two b128 loads.
            h8 x = *(const h8*)(pA + kc * 32 + 16 * hi);
            h8 y = *(const h8*)(pA + kc * 32 + 16 * hi + 8);
            v16h af = __builtin_shufflevector(x, y, 0, 1, 2, 3, 4, 5, 6, 7,
                                              8, 9, 10, 11, 12, 13, 14, 15);
            acc0 = __builtin_amdgcn_wmma_f32_16x16x32_f16(
                false, af, false, bfrag0[kc], (short)0, acc0, false, false);
            acc1 = __builtin_amdgcn_wmma_f32_16x16x32_f16(
                false, af, false, bfrag1[kc], (short)0, acc1, false, false);
        }

        // Online softmax update: 8 rows per accumulator, one column each.
        {
            float tmax = acc0[0];
#pragma unroll
            for (int v = 1; v < 8; ++v) tmax = fmaxf(tmax, acc0[v]);
            const float nm = fmaxf(m0, tmax);
            float add = 0.f;
#pragma unroll
            for (int v = 0; v < 8; ++v) add += __expf(acc0[v] - nm);
            s0 = s0 * __expf(m0 - nm) + add;
            m0 = nm;
        }
        {
            float tmax = acc1[0];
#pragma unroll
            for (int v = 1; v < 8; ++v) tmax = fmaxf(tmax, acc1[v]);
            const float nm = fmaxf(m1, tmax);
            float add = 0.f;
#pragma unroll
            for (int v = 0; v < 8; ++v) add += __expf(acc1[v] - nm);
            s1 = s1 * __expf(m1 - nm) + add;
            m1 = nm;
        }
    }

    // Combine 8 waves x 2 lane-halves per column, then logsumexp.
    // Column set t in {0,1}: global column = j0 + 16*t + (lane&15).
    __shared__ float lds_m[2][256];
    __shared__ float lds_s[2][256];
    __shared__ float colval[32];
    lds_m[0][tid] = m0;
    lds_s[0][tid] = s0;
    lds_m[1][tid] = m1;
    lds_s[1][tid] = s1;
    __syncthreads();

    if (tid < 32) {
        const int t = tid >> 4;       // which 16-column tile
        const int c = tid & 15;       // column within tile
        float M = -1e30f;
#pragma unroll
        for (int e = 0; e < 16; ++e) {
            const int idx = (e >> 1) * 32 + (e & 1) * 16 + c;
            M = fmaxf(M, lds_m[t][idx]);
        }
        float S = 0.f;
#pragma unroll
        for (int e = 0; e < 16; ++e) {
            const int idx = (e >> 1) * 32 + (e & 1) * 16 + c;
            S += lds_s[t][idx] * __expf(lds_m[t][idx] - M);
        }
        const float lse = M + __logf(S);
        colval[tid] = lse - diag[j0 + t * 16 + c];
    }
    __syncthreads();

    if (tid == 0) {
        float t = 0.f;
#pragma unroll
        for (int c = 0; c < 32; ++c) t += colval[c];
        partial[blockIdx.x] = t;
    }
}

// ---------------------------------------------------------------------------
// Phase 3: deterministic tree-reduce of 256 per-block partials.
// ---------------------------------------------------------------------------
__global__ void __launch_bounds__(256)
reduce_kernel(const float* __restrict__ partial, float* __restrict__ out)
{
    __shared__ float buf[256];
    buf[threadIdx.x] = partial[threadIdx.x];
    __syncthreads();
#pragma unroll
    for (int s = 128; s > 0; s >>= 1) {
        if (threadIdx.x < s) buf[threadIdx.x] += buf[threadIdx.x + s];
        __syncthreads();
    }
    if (threadIdx.x == 0) out[0] = buf[0];
}

// ---------------------------------------------------------------------------
extern "C" void kernel_launch(void* const* d_in, const int* in_sizes, int n_in,
                              void* d_out, int out_size, void* d_ws, size_t ws_size,
                              hipStream_t stream)
{
    const float* e1 = (const float*)d_in[0];   // encoder_embedding1 (anchors)
    const float* e2 = (const float*)d_in[1];   // encoder_embedding2 (contrast)

    _Float16* aH      = (_Float16*)d_ws;                       // 4 MB
    _Float16* cH      = aH + (size_t)B_ROWS * DDIM;            // 4 MB
    float*    diag    = (float*)(cH + (size_t)B_ROWS * DDIM);  // 32 KB
    float*    partial = diag + B_ROWS;                         // 1 KB

    normalize_kernel<<<B_ROWS / 8, 256, 0, stream>>>(e1, e2, aH, cH, diag);
    simloss_kernel<<<B_ROWS / 32, 256, 0, stream>>>(aH, cH, diag, partial);
    reduce_kernel<<<1, 256, 0, stream>>>(partial, (float*)d_out);
}
